// aggregator_21887153340714
// MI455X (gfx1250) — matrix operand order repaired and checked by reading, not compiled, not run on hardware
//
#include <hip/hip_runtime.h>

typedef _Float16 v16h __attribute__((ext_vector_type(16)));
typedef _Float16 v8h  __attribute__((ext_vector_type(8)));
typedef _Float16 v2h  __attribute__((ext_vector_type(2)));
typedef float    v8f  __attribute__((ext_vector_type(8)));

#define NUSERS 50000
#define WAVES_PER_BLOCK 8

static __device__ inline v16h cat16(v8h lo, v8h hi) {
  v16h r;
#pragma unroll
  for (int i = 0; i < 8; ++i) { r[i] = lo[i]; r[i + 8] = hi[i]; }
  return r;
}

// A-tile (16x32 f16, row-major source). base = tile origin, stride in elements.
// ISA layout: lane<16 -> row=lane, K {0..7, 16..23}; lane>=16 -> row=lane-16, K {8..15, 24..31}
static __device__ inline v16h load_a_tile(const _Float16* base, int stride, int lane) {
  const int row  = lane & 15;
  const int koff = (lane < 16) ? 0 : 8;
  const _Float16* p = base + row * stride + koff;
  v8h lo = *(const v8h*)(p);
  v8h hi = *(const v8h*)(p + 16);
  return cat16(lo, hi);
}

// A-tile where every row is identical (broadcast node feature): row ignored.
static __device__ inline v16h load_a_bcast(const _Float16* base, int lane) {
  const int koff = (lane < 16) ? 0 : 8;
  v8h lo = *(const v8h*)(base + koff);
  v8h hi = *(const v8h*)(base + koff + 16);
  return cat16(lo, hi);
}

// B-tile (32x16 f16) from transposed weights Wt[n][k] (row-major, stride=K).
// ISA layout: lane<16 -> col=lane, K {0..7, 16..23}; lane>=16 -> col=lane-16, K {8..15, 24..31}
static __device__ inline v16h load_b_tile(const _Float16* base, int stride, int lane) {
  const int col   = lane & 15;
  const int kbase = (lane < 16) ? 0 : 8;
  const _Float16* p = base + col * stride + kbase;
  v8h lo = *(const v8h*)(p);
  v8h hi = *(const v8h*)(p + 16);
  return cat16(lo, hi);
}

static __device__ inline v8f wmma_f16(v16h a, v16h b, v8f c) {
  return __builtin_amdgcn_wmma_f32_16x16x32_f16(false, a, false, b, (short)0, c,
                                                false, false);
}

static __device__ inline void lds_fence() {
  asm volatile("s_wait_dscnt 0" ::: "memory");
}

__global__ __launch_bounds__(256) void graphrec_agg_kernel(
    const int*   __restrict__ ui,     // [N, 32]
    const float* __restrict__ ufeat,  // [N, 64]
    const float* __restrict__ ifeat,  // [M, 64]
    const float* __restrict__ w1,     // [128, 64]
    const float* __restrict__ b1,     // [64]
    const float* __restrict__ w2,     // [64, 64]
    const float* __restrict__ b2,     // [64]
    const float* __restrict__ w3,     // [64]
    float*       __restrict__ out)    // [N, 64]
{
  __shared__ alignas(16) _Float16 sW1t[64 * 128];                 // W1^T f16
  __shared__ alignas(16) _Float16 sW2t[64 * 64];                  // W2^T f16
  __shared__ alignas(16) _Float16 sW3h[64];
  __shared__ alignas(16) float    sB1[64];
  __shared__ alignas(16) float    sB2[64];
  __shared__ alignas(16) _Float16 sXg[WAVES_PER_BLOCK][32 * 64];  // neighbor f16
  __shared__ alignas(16) _Float16 sXn[WAVES_PER_BLOCK][64];       // node f16 (bcast row)
  __shared__ alignas(16) _Float16 sH1[WAVES_PER_BLOCK][32 * 64];
  __shared__ alignas(16) _Float16 sH2[WAVES_PER_BLOCK][32 * 64];
  __shared__ alignas(16) float    sAtt[WAVES_PER_BLOCK][32];
  __shared__ alignas(16) int      sIdx[WAVES_PER_BLOCK][32];

  const int tid  = threadIdx.x;
  const int lane = tid & 31;
  const int wid  = tid >> 5;
  const int user = blockIdx.x * WAVES_PER_BLOCK + wid;

  // ---- stage weights (transposed) + biases into LDS, f16 ----
  for (int e = tid; e < 64 * 128; e += 256) {
    const int n = e >> 7, k = e & 127;
    sW1t[e] = (_Float16)w1[k * 64 + n];
  }
  for (int e = tid; e < 64 * 64; e += 256) {
    const int n = e >> 6, k = e & 63;
    sW2t[e] = (_Float16)w2[k * 64 + n];
  }
  if (tid < 64) {
    sW3h[tid] = (_Float16)w3[tid];
    sB1[tid]  = b1[tid];
    sB2[tid]  = b2[tid];
  }
  __syncthreads();

  _Float16* Xg  = sXg[wid];
  _Float16* Xn  = sXn[wid];
  _Float16* H1  = sH1[wid];
  _Float16* H2  = sH2[wid];
  float*    Att = sAtt[wid];
  int*      Idx = sIdx[wid];

  // ---- gather: indices, node feature, neighbor rows (coalesced) ----
  Idx[lane] = ui[user * 32 + lane];
  {
    const float2 nf = *(const float2*)(ufeat + user * 64 + lane * 2);
    v2h p; p[0] = (_Float16)nf.x; p[1] = (_Float16)nf.y;
    *(v2h*)(Xn + lane * 2) = p;
  }
  lds_fence();
#pragma unroll 4
  for (int l = 0; l < 32; ++l) {
    const int idx = Idx[l];
    const float2 v = *(const float2*)(ifeat + idx * 64 + lane * 2);
    v2h p; p[0] = (_Float16)v.x; p[1] = (_Float16)v.y;
    *(v2h*)(Xg + l * 64 + lane * 2) = p;
  }
  lds_fence();

  // ---- layer 1: H1 = relu([Xg|Xn] @ W1 + b1), M=32 K=128 N=64 ----
#pragma unroll
  for (int mt = 0; mt < 2; ++mt) {
#pragma unroll
    for (int nt = 0; nt < 4; ++nt) {
      v8f c = {};
#pragma unroll
      for (int kt = 0; kt < 4; ++kt) {
        const v16h a = (kt < 2) ? load_a_tile(Xg + mt * 16 * 64 + kt * 32, 64, lane)
                                : load_a_bcast(Xn + (kt - 2) * 32, lane);
        const v16h b = load_b_tile(sW1t + nt * 16 * 128 + kt * 32, 128, lane);
        c = wmma_f16(a, b, c);
      }
      const int   n     = nt * 16 + (lane & 15);
      const float bn    = sB1[n];
      const int   mbase = mt * 16 + ((lane < 16) ? 0 : 8);
#pragma unroll
      for (int r = 0; r < 8; ++r) {
        float v = c[r] + bn;
        v = v > 0.f ? v : 0.f;
        H1[(mbase + r) * 64 + n] = (_Float16)v;
      }
    }
  }
  lds_fence();

  // ---- layer 2: H2 = relu(H1 @ W2 + b2), M=32 K=64 N=64 ----
#pragma unroll
  for (int mt = 0; mt < 2; ++mt) {
#pragma unroll
    for (int nt = 0; nt < 4; ++nt) {
      v8f c = {};
#pragma unroll
      for (int kt = 0; kt < 2; ++kt) {
        const v16h a = load_a_tile(H1 + mt * 16 * 64 + kt * 32, 64, lane);
        const v16h b = load_b_tile(sW2t + nt * 16 * 64 + kt * 32, 64, lane);
        c = wmma_f16(a, b, c);
      }
      const int   n     = nt * 16 + (lane & 15);
      const float bn    = sB2[n];
      const int   mbase = mt * 16 + ((lane < 16) ? 0 : 8);
#pragma unroll
      for (int r = 0; r < 8; ++r) {
        float v = c[r] + bn;
        v = v > 0.f ? v : 0.f;
        H2[(mbase + r) * 64 + n] = (_Float16)v;
      }
    }
  }
  lds_fence();

  // ---- layer 3: per-lane logit for its own neighbor row, then wave softmax ----
  float logit = 0.f;
#pragma unroll
  for (int j = 0; j < 64; j += 8) {
    const v8h hrow = *(const v8h*)(H2 + lane * 64 + j);
    const v8h wrow = *(const v8h*)(sW3h + j);
#pragma unroll
    for (int u = 0; u < 8; ++u) logit += (float)hrow[u] * (float)wrow[u];
  }
  float mx = logit;
#pragma unroll
  for (int off = 16; off >= 1; off >>= 1) mx = fmaxf(mx, __shfl_xor(mx, off, 32));
  const float e = __expf(logit - mx);
  float s = e;
#pragma unroll
  for (int off = 16; off >= 1; off >>= 1) s += __shfl_xor(s, off, 32);
  Att[lane] = e / s;
  lds_fence();

  // ---- attention-weighted sum of neighbor features; each lane owns 2 dims ----
  float a0 = 0.f, a1 = 0.f;
#pragma unroll 4
  for (int l = 0; l < 32; ++l) {
    const float w = Att[l];
    const v2h   p = *(const v2h*)(Xg + l * 64 + lane * 2);
    a0 += w * (float)p[0];
    a1 += w * (float)p[1];
  }
  float2 o; o.x = a0; o.y = a1;
  *(float2*)(out + user * 64 + lane * 2) = o;
}

extern "C" void kernel_launch(void* const* d_in, const int* in_sizes, int n_in,
                              void* d_out, int out_size, void* d_ws, size_t ws_size,
                              hipStream_t stream) {
  (void)in_sizes; (void)n_in; (void)out_size; (void)d_ws; (void)ws_size;
  const int*   ui    = (const int*)d_in[0];
  const float* ufeat = (const float*)d_in[1];
  const float* ifeat = (const float*)d_in[2];
  const float* w1    = (const float*)d_in[3];
  const float* b1    = (const float*)d_in[4];
  const float* w2    = (const float*)d_in[5];
  const float* b2    = (const float*)d_in[6];
  const float* w3    = (const float*)d_in[7];
  // d_in[8] = att_b3: uniform logit shift, softmax-invariant -> unused
  float* out = (float*)d_out;

  const dim3 grid(NUSERS / WAVES_PER_BLOCK);  // 6250 blocks, 8 users each
  graphrec_agg_kernel<<<grid, 256, 0, stream>>>(ui, ufeat, ifeat, w1, b1, w2, b2,
                                                w3, out);
}
